// PointAugment_86672440033979
// MI455X (gfx1250) — matrix operand order repaired and checked by reading, not compiled, not run on hardware
//
#include <hip/hip_runtime.h>
#include <hip/hip_bf16.h>

typedef __attribute__((ext_vector_type(16))) _Float16 v16h;
typedef __attribute__((ext_vector_type(8)))  _Float16 v8h;
typedef __attribute__((ext_vector_type(8)))  float    v8f;
typedef __attribute__((ext_vector_type(4)))  float    v4f;

// ---------------------------------------------------------------------------
// Kernel A: raising MLP  F = relu(relu(pos@Wr0+br0)@Wr1+br1)  -> F_f16 [N,64]
// One wave handles 16 points per tile; layer0 in VALU (each lane computes the
// 16 hidden values its WMMA A-slot needs), layer1 = 4x v_wmma_f32_16x16x32_f16.
// Results staged through wave-private LDS so global stores are 4x b128 / lane.
// ---------------------------------------------------------------------------
__global__ __launch_bounds__(256) void pa_raise_kernel(
    const float* __restrict__ pos, const float* __restrict__ Wr0,
    const float* __restrict__ br0, const float* __restrict__ Wr1,
    const float* __restrict__ br1, _Float16* __restrict__ Ff16,
    int ntiles)
{
  __shared__ __align__(16) _Float16 stage[8][16 * 72];  // padded rows

  const int lane  = threadIdx.x & 31;
  const int wave  = threadIdx.x >> 5;
  const int hi    = lane >> 4;
  const int m     = lane & 15;
  const int base8 = hi * 8;

  // Persistent per-lane weights (registers).
  float w0r[3][16];
  float b0r[16];
#pragma unroll
  for (int i = 0; i < 16; ++i) {
    const int k = (i < 8) ? (base8 + i) : (16 + base8 + (i - 8));
    b0r[i] = br0[k];
#pragma unroll
    for (int r = 0; r < 3; ++r) w0r[r][i] = Wr0[r * 32 + k];
  }
  // B operand: lane l holds Wr1 column n=l%16 over the same k order as A.
  v16h Bv[4];
#pragma unroll
  for (int nt = 0; nt < 4; ++nt)
#pragma unroll
    for (int i = 0; i < 16; ++i)
      Bv[nt][i] = (_Float16)Wr1[((i < 8) ? (base8 + i) : (16 + base8 + (i - 8))) * 64 + nt * 16 + m];
  float b1r[4];
#pragma unroll
  for (int nt = 0; nt < 4; ++nt) b1r[nt] = br1[nt * 16 + m];

  for (int tile = blockIdx.x * 8 + wave; tile < ntiles; tile += gridDim.x * 8) {
    const int p = tile * 16 + m;
    const float px = pos[p * 3 + 0];
    const float py = pos[p * 3 + 1];
    const float pz = pos[p * 3 + 2];

    v16h A;
#pragma unroll
    for (int i = 0; i < 16; ++i) {
      float h = b0r[i] + px * w0r[0][i] + py * w0r[1][i] + pz * w0r[2][i];
      h = h > 0.f ? h : 0.f;
      A[i] = (_Float16)h;
    }
#pragma unroll
    for (int nt = 0; nt < 4; ++nt) {
      v8f c = {};
      c = __builtin_amdgcn_wmma_f32_16x16x32_f16(
          false, A, false, Bv[nt], (short)0, c, false, false);
#pragma unroll
      for (int r = 0; r < 8; ++r) {
        float f = c[r] + b1r[nt];
        f = f > 0.f ? f : 0.f;
        stage[wave][(r + 8 * hi) * 72 + nt * 16 + m] = (_Float16)f;
      }
    }
    // wave-private staging: split-counter wait instead of a full barrier
    asm volatile("s_wait_dscnt 0" ::: "memory");

    // Coalesced writeback: lane stores 64 contiguous bytes (half an F row).
    {
      const int row  = lane >> 1;
      const int half = lane & 1;
      const _Float16* sp = &stage[wave][row * 72 + half * 32];
      _Float16* dp = Ff16 + (size_t)(tile * 16 + row) * 64 + half * 32;
#pragma unroll
      for (int i = 0; i < 4; ++i)
        *(v8h*)(dp + i * 8) = *(const v8h*)(sp + i * 8);
    }
  }
}

// ---------------------------------------------------------------------------
// Kernel B: segment offsets via binary search over the sorted batch array.
// off[t] = first index i with batch[i] >= t, t in [0, B].
// ---------------------------------------------------------------------------
__global__ void pa_offsets_kernel(const int* __restrict__ batch,
                                  int* __restrict__ off, int N, int B)
{
  const int t = blockIdx.x * blockDim.x + threadIdx.x;
  if (t > B) return;
  int lo = 0, hi = N;
  while (lo < hi) {
    const int mid = (lo + hi) >> 1;
    if (batch[mid] < t) lo = mid + 1; else hi = mid;
  }
  off[t] = lo;
}

// ---------------------------------------------------------------------------
// Kernel C: per-cloud mean of F  -> G (fp32 + f16 copies). One block / cloud.
// ---------------------------------------------------------------------------
__global__ __launch_bounds__(256) void pa_segmean_kernel(
    const _Float16* __restrict__ Ff16, const int* __restrict__ off,
    float* __restrict__ Gf32, _Float16* __restrict__ Gf16, int B)
{
  __shared__ float red[256];
  const int b = blockIdx.x;
  const int start = off[b], end = off[b + 1];
  const int col = threadIdx.x & 63;
  const int seg = threadIdx.x >> 6;
  float s = 0.f;
  for (int r = start + seg; r < end; r += 4)
    s += (float)Ff16[(size_t)r * 64 + col];
  red[threadIdx.x] = s;
  __syncthreads();
  if (threadIdx.x < 64) {
    const float tot = red[threadIdx.x] + red[threadIdx.x + 64] +
                      red[threadIdx.x + 128] + red[threadIdx.x + 192];
    float cnt = (float)(end - start);
    cnt = cnt > 1.f ? cnt : 1.f;           // matches max(counts, 1.0)
    const float mean = tot / cnt;
    Gf32[(size_t)b * 64 + col] = mean;
    Gf16[(size_t)b * 64 + col] = (_Float16)mean;
  }
}

// ---------------------------------------------------------------------------
// Kernel D: rotation head (tiny): M = relu([noise_rot|G]@Wm0+bm0)@Wm1+bm1.
// 4 clouds per block, 64 threads per cloud.
// ---------------------------------------------------------------------------
__global__ __launch_bounds__(256) void pa_rot_kernel(
    const float* __restrict__ noise_rot, const float* __restrict__ Gf32,
    const float* __restrict__ Wm0, const float* __restrict__ bm0,
    const float* __restrict__ Wm1, const float* __restrict__ bm1,
    float* __restrict__ Mbuf, int B)
{
  __shared__ float feat[4][128];
  __shared__ float hsh[4][64];
  const int sub = threadIdx.x >> 6;
  const int j   = threadIdx.x & 63;
  int b = blockIdx.x * 4 + sub;
  if (b >= B) b = B - 1;                    // clamp: keeps barriers uniform
  feat[sub][j]      = noise_rot[(size_t)b * 64 + j];
  feat[sub][64 + j] = Gf32[(size_t)b * 64 + j];
  __syncthreads();
  float acc = bm0[j];
  for (int i = 0; i < 128; ++i) acc += feat[sub][i] * Wm0[i * 64 + j];
  hsh[sub][j] = acc > 0.f ? acc : 0.f;
  __syncthreads();
  if (j < 9) {
    float a = bm1[j];
    for (int i = 0; i < 64; ++i) a += hsh[sub][i] * Wm1[i * 9 + j];
    Mbuf[(size_t)b * 9 + j] = a;
  }
}

// ---------------------------------------------------------------------------
// Kernel E: translation head + output.
// feat = [F | G[batch] | noise_tr]  (N x 192, f16)
// hidden = relu(feat @ Wt0 + bt0);  T = hidden @ Wt1 + bt1
// out = T + pos @ M[batch]
// Wt0 cached in LDS as f16 (row stride 72 halves); 24 WMMAs / 16 points.
// ---------------------------------------------------------------------------
__global__ __launch_bounds__(256) void pa_trans_kernel(
    const _Float16* __restrict__ Ff16, const _Float16* __restrict__ Gf16,
    const int* __restrict__ batch, const float* __restrict__ noise,
    const float* __restrict__ pos, const float* __restrict__ Mbuf,
    const float* __restrict__ Wt0, const float* __restrict__ bt0,
    const float* __restrict__ Wt1, const float* __restrict__ bt1,
    float* __restrict__ out, int ntiles)
{
  __shared__ __align__(16) _Float16 w0[192 * 72];   // padded rows: bank spread
  __shared__ float w1[64 * 3];
  __shared__ float hid[8][16 * 65];                 // stride 65: conflict-free

  for (int idx = threadIdx.x; idx < 192 * 64; idx += 256)
    w0[(idx >> 6) * 72 + (idx & 63)] = (_Float16)Wt0[idx];
  for (int idx = threadIdx.x; idx < 192; idx += 256)
    w1[idx] = Wt1[idx];
  __syncthreads();

  const int lane  = threadIdx.x & 31;
  const int wave  = threadIdx.x >> 5;
  const int hi    = lane >> 4;
  const int m     = lane & 15;
  const int base8 = hi * 8;

  float bt0r[4];
#pragma unroll
  for (int nt = 0; nt < 4; ++nt) bt0r[nt] = bt0[nt * 16 + m];
  const float t1x = bt1[0], t1y = bt1[1], t1z = bt1[2];

  for (int tile = blockIdx.x * 8 + wave; tile < ntiles; tile += gridDim.x * 8) {
    const int p = tile * 16 + m;
    const int b = batch[p];

    v8f c[4] = {};
#pragma unroll
    for (int kt = 0; kt < 6; ++kt) {
      const int cc = (kt & 1) * 32 + base8;   // column of run1 in 64-col source
      v16h A;
      if (kt < 4) {
        const _Float16* src = (kt < 2) ? (Ff16 + (size_t)p * 64)
                                       : (Gf16 + (size_t)b * 64);
        const v8h lo = *(const v8h*)(src + cc);
        const v8h hh = *(const v8h*)(src + cc + 16);
#pragma unroll
        for (int i = 0; i < 8; ++i) { A[i] = lo[i]; A[8 + i] = hh[i]; }
      } else {
        const float* src = noise + (size_t)p * 64 + cc;
        const v4f f0 = *(const v4f*)(src + 0);
        const v4f f1 = *(const v4f*)(src + 4);
        const v4f f2 = *(const v4f*)(src + 16);
        const v4f f3 = *(const v4f*)(src + 20);
#pragma unroll
        for (int i = 0; i < 4; ++i) {
          A[i]      = (_Float16)f0[i];
          A[4 + i]  = (_Float16)f1[i];
          A[8 + i]  = (_Float16)f2[i];
          A[12 + i] = (_Float16)f3[i];
        }
      }
#pragma unroll
      for (int nt = 0; nt < 4; ++nt) {
        const _Float16* bp = &w0[(kt * 32) * 72 + nt * 16 + m];
        v16h Bv;
#pragma unroll
        for (int i = 0; i < 8; ++i) { Bv[i] = bp[(base8 + i) * 72]; Bv[8 + i] = bp[(16 + base8 + i) * 72]; }
        c[nt] = __builtin_amdgcn_wmma_f32_16x16x32_f16(
            false, A, false, Bv, (short)0, c[nt], false, false);
      }
    }

    // bias + relu -> wave-private LDS staging (C layout -> row-major)
#pragma unroll
    for (int nt = 0; nt < 4; ++nt) {
#pragma unroll
      for (int r = 0; r < 8; ++r) {
        float v = c[nt][r] + bt0r[nt];
        v = v > 0.f ? v : 0.f;
        hid[wave][(r + 8 * hi) * 65 + nt * 16 + m] = v;
      }
    }
    // wave-private staging: split-counter wait instead of a full barrier
    asm volatile("s_wait_dscnt 0" ::: "memory");

    if (lane < 16) {
      float a0 = t1x, a1 = t1y, a2 = t1z;
#pragma unroll 8
      for (int n2 = 0; n2 < 64; ++n2) {
        const float hv = hid[wave][lane * 65 + n2];
        a0 += hv * w1[n2 * 3 + 0];
        a1 += hv * w1[n2 * 3 + 1];
        a2 += hv * w1[n2 * 3 + 2];
      }
      const float px = pos[p * 3 + 0];
      const float py = pos[p * 3 + 1];
      const float pz = pos[p * 3 + 2];
      const float* Mr = Mbuf + (size_t)b * 9;
      out[p * 3 + 0] = a0 + px * Mr[0] + py * Mr[3] + pz * Mr[6];
      out[p * 3 + 1] = a1 + px * Mr[1] + py * Mr[4] + pz * Mr[7];
      out[p * 3 + 2] = a2 + px * Mr[2] + py * Mr[5] + pz * Mr[8];
    }
  }
}

// ---------------------------------------------------------------------------
extern "C" void kernel_launch(void* const* d_in, const int* in_sizes, int n_in,
                              void* d_out, int out_size, void* d_ws, size_t ws_size,
                              hipStream_t stream) {
  (void)n_in; (void)out_size; (void)ws_size;
  const float* pos       = (const float*)d_in[0];
  const int*   batch     = (const int*)d_in[1];
  const float* noise_rot = (const float*)d_in[2];
  const float* noise_tr  = (const float*)d_in[3];
  const float* Wr0 = (const float*)d_in[4];
  const float* br0 = (const float*)d_in[5];
  const float* Wr1 = (const float*)d_in[6];
  const float* br1 = (const float*)d_in[7];
  const float* Wm0 = (const float*)d_in[8];
  const float* bm0 = (const float*)d_in[9];
  const float* Wm1 = (const float*)d_in[10];
  const float* bm1 = (const float*)d_in[11];
  const float* Wt0 = (const float*)d_in[12];
  const float* bt0 = (const float*)d_in[13];
  const float* Wt1 = (const float*)d_in[14];
  const float* bt1 = (const float*)d_in[15];
  float* out = (float*)d_out;

  const int N = in_sizes[0] / 3;
  const int B = in_sizes[2] / 64;
  const int ntiles = (N + 15) / 16;

  // Workspace carve-up.
  char* ws = (char*)d_ws;
  _Float16* Ff16 = (_Float16*)ws;           ws += (size_t)N * 64 * sizeof(_Float16);
  _Float16* Gf16 = (_Float16*)ws;           ws += (size_t)B * 64 * sizeof(_Float16);
  float*    Gf32 = (float*)ws;              ws += (size_t)B * 64 * sizeof(float);
  float*    Mbuf = (float*)ws;              ws += (size_t)B * 9  * sizeof(float);
  int*      off  = (int*)ws;                ws += (size_t)(B + 1) * sizeof(int);

  int gA = (ntiles + 7) / 8; if (gA > 640) gA = 640;
  pa_raise_kernel<<<gA, 256, 0, stream>>>(pos, Wr0, br0, Wr1, br1, Ff16, ntiles);

  const int gOff = (B + 1 + 255) / 256;
  pa_offsets_kernel<<<gOff, 256, 0, stream>>>(batch, off, N, B);

  pa_segmean_kernel<<<B, 256, 0, stream>>>(Ff16, off, Gf32, Gf16, B);

  pa_rot_kernel<<<(B + 3) / 4, 256, 0, stream>>>(noise_rot, Gf32, Wm0, bm0,
                                                 Wm1, bm1, Mbuf, B);

  int gE = (ntiles + 7) / 8; if (gE > 768) gE = 768;
  pa_trans_kernel<<<gE, 256, 0, stream>>>(Ff16, Gf16, batch, noise_tr, pos,
                                          Mbuf, Wt0, bt0, Wt1, bt1, out, ntiles);
}
